// mLSTMLayer_61967788147076
// MI455X (gfx1250) — compile-verified, matrix operand order checked by reading, not run on hardware
//
#include <hip/hip_runtime.h>
#include <hip/hip_bf16.h>
#include <stdint.h>

typedef __bf16 bf16;
typedef __attribute__((ext_vector_type(16))) __bf16 v16bf;
typedef __attribute__((ext_vector_type(8)))  float  v8f;

#define LPAD 40   // LDS row stride in halves: 32 data + 8 pad (64B data + 16B pad, 16B aligned)

// ---- Tensor Data Mover support (probe-confirmed builtin; guarded fallback) ----
#if defined(__has_builtin)
#if __has_builtin(__builtin_amdgcn_tensor_load_to_lds) && __has_builtin(__builtin_amdgcn_s_wait_tensorcnt)
#define USE_TDM 1
#endif
#endif
#ifndef USE_TDM
#define USE_TDM 0
#endif

#if USE_TDM
typedef __attribute__((ext_vector_type(4))) unsigned int v4u;
typedef __attribute__((ext_vector_type(4))) int v4i;
typedef __attribute__((ext_vector_type(8))) int v8i;

// Generic pointer to a __shared__ object: low 32 bits of the flat address are the
// workgroup-relative LDS byte offset (LDS aperture truncation rule).
__device__ __forceinline__ unsigned ldsoff(const void* p) {
  return (unsigned)(unsigned long long)p;
}

// 2-D bf16 tile load: 'rows' rows of 32 elements, row stride row_elems (elements),
// into LDS with 16B padding after every 64B row chunk -> matches LPAD layout.
__device__ __forceinline__ void tdm_load_2d(unsigned lds_byte, const void* g,
                                            unsigned row_elems, unsigned rows) {
  unsigned long long ga = (unsigned long long)g;
  v4u g0; v8i g1; v4i g2; v4i g3;
  g0[0] = 1u;                                           // count=1, user descriptor
  g0[1] = lds_byte;                                     // lds_addr
  g0[2] = (unsigned)ga;                                 // global_addr lo
  g0[3] = (unsigned)((ga >> 32) & 0x1ffffffu) | 0x80000000u;  // global_addr hi | type=2
  // data_size=2B, pad_enable, pad_interval=16 DWORDs (64B), pad_amount=4 DWORDs (16B)
  g1[0] = (int)((1u << 16) | (1u << 20) | (3u << 22) | (3u << 25));
  g1[1] = (int)((row_elems & 0xffffu) << 16);           // tensor_dim0 lo16
  g1[2] = (int)((row_elems >> 16) & 0xffffu);           // tensor_dim0 hi16 | tensor_dim1 lo16(=0)
  g1[3] = (int)(16u | (32u << 16));                     // tensor_dim1 hi16 (dim1=1<<20) | tile_dim0=32
  g1[4] = (int)(rows & 0xffffu);                        // tile_dim1=rows, tile_dim2=0
  g1[5] = (int)row_elems;                               // tensor_dim0_stride lo32 (elements)
  g1[6] = 0; g1[7] = 0;
  g2[0] = g2[1] = g2[2] = g2[3] = 0; g3 = g2;
#if __clang_major__ >= 23
  v8i g4; g4[0]=g4[1]=g4[2]=g4[3]=g4[4]=g4[5]=g4[6]=g4[7]=0;
  __builtin_amdgcn_tensor_load_to_lds(g0, g1, g2, g3, g4, 0);
#else
  __builtin_amdgcn_tensor_load_to_lds(g0, g1, g2, g3, 0);
#endif
}
#endif  // USE_TDM

union FragU { uint4 u[2]; v16bf v; };

__device__ __forceinline__ v8f wmma_bf16(const FragU& a, const FragU& b, v8f c) {
  return __builtin_amdgcn_wmma_f32_16x16x32_bf16(false, a.v, false, b.v, (short)0, c, false, false);
}

// Fragment load from LDS tile (row-major [rows][LPAD]); 16-bit A-layout:
// lane<16 -> k base 0, lane>=16 -> k base 8; VGPR0..3 = k kb..kb+7, VGPR4..7 = kb+16..kb+23.
__device__ __forceinline__ void lds_frag(const bf16* l, int rbase, int lane, FragU& f) {
  int r  = rbase + (lane & 15);
  int kb = (lane >> 4) << 3;
  const bf16* p = l + r * LPAD + kb;
  f.u[0] = *(const uint4*)p;
  f.u[1] = *(const uint4*)(p + 16);
}

// Stage rows x 32 bf16 tile (row-major source) into LDS (TDM or fallback copy).
__device__ __forceinline__ void stage_rows(bf16* l, const bf16* src, long ld, long r0, long k0,
                                           int rows, int tid) {
#if USE_TDM
  if (tid < 32) tdm_load_2d(ldsoff(l), src + r0 * ld + k0, (unsigned)ld, (unsigned)rows);
#else
  int chunks = rows * 4;
  for (int i = tid; i < chunks; i += 256) {
    int r = i >> 2, c8 = (i & 3) << 3;
    *(uint4*)(l + r * LPAD + c8) = *(const uint4*)(src + (r0 + r) * ld + (k0 + c8));
  }
#endif
}

// Complete staging: TDM wait (issuing wave) + workgroup barrier.
__device__ __forceinline__ void stage_commit(int tid) {
#if USE_TDM
  if (tid < 32) __builtin_amdgcn_s_wait_tensorcnt(0);
#endif
  __syncthreads();
}

// Stage 64x32 B^T tile from row-major (k,n): coalesced b128 loads + 8 scalar LDS stores.
__device__ __forceinline__ void stage_B64_trans(bf16* l, const bf16* Bsrc, long ldb, long k0, long n0,
                                                int tid) {
  int kk = tid >> 3;          // 0..31
  int nc = (tid & 7) << 3;    // 0,8,...,56
  const bf16* src = Bsrc + (k0 + kk) * ldb + n0 + nc;
  __builtin_prefetch((const void*)(src + 32 * ldb), 0, 0);   // next k-tile
  union { bf16 h[8]; uint4 u; } t;
  t.u = *(const uint4*)src;
#pragma unroll
  for (int j = 0; j < 8; ++j) l[(nc + j) * LPAD + kk] = t.h[j];
}

// ---------------- generic bf16 WMMA GEMM: C(fp32) = A @ B, B row-major (k,n) ----------------
__global__ void __launch_bounds__(256) k_gemm_bt(const bf16* __restrict__ A, const bf16* __restrict__ B,
                                                 float* __restrict__ C, int K, int lda, int ldb, int ldc) {
  __shared__ bf16 lA[128 * LPAD];
  __shared__ bf16 lB[64 * LPAD];
  int tid = threadIdx.x, lane = tid & 31, wid = tid >> 5;
  int wm = (wid >> 1) * 32, wn = (wid & 1) * 32;
  long m0 = (long)blockIdx.y * 128, n0 = (long)blockIdx.x * 64;
  v8f acc[2][2] = {};
  for (int k0 = 0; k0 < K; k0 += 32) {
    __syncthreads();
    stage_rows(lA, A, lda, m0, k0, 128, tid);
    stage_B64_trans(lB, B, ldb, k0, n0, tid);
    stage_commit(tid);
    FragU a0, a1, b0, b1;
    lds_frag(lA, wm, lane, a0);  lds_frag(lA, wm + 16, lane, a1);
    lds_frag(lB, wn, lane, b0);  lds_frag(lB, wn + 16, lane, b1);
    acc[0][0] = wmma_bf16(a0, b0, acc[0][0]);
    acc[0][1] = wmma_bf16(a0, b1, acc[0][1]);
    acc[1][0] = wmma_bf16(a1, b0, acc[1][0]);
    acc[1][1] = wmma_bf16(a1, b1, acc[1][1]);
  }
  int nl = lane & 15, hi = lane >> 4;
#pragma unroll
  for (int mi = 0; mi < 2; ++mi)
#pragma unroll
    for (int ni = 0; ni < 2; ++ni)
#pragma unroll
      for (int r = 0; r < 8; ++r)
        C[(m0 + wm + mi * 16 + r + 8 * hi) * (long)ldc + n0 + wn + ni * 16 + nl] = acc[mi][ni][r];
}

// ---------------- masked, gated q@k^T with rowsum accumulation ----------------
__global__ void __launch_bounds__(256) k_qk(const bf16* __restrict__ q, const bf16* __restrict__ kmat,
                                            bf16* __restrict__ Cb, float* __restrict__ rowsum,
                                            const float* __restrict__ G, const float* __restrict__ Mx) {
  long bh = blockIdx.z;
  long s0 = (long)blockIdx.y * 128, t0 = (long)blockIdx.x * 64;
  if (t0 > s0 + 127) return;  // fully above diagonal: never read by PV
  __shared__ bf16 lA[128 * LPAD];
  __shared__ bf16 lB[64 * LPAD];
  __shared__ float sM[128];
  __shared__ float sG[64];
  int tid = threadIdx.x, lane = tid & 31, wid = tid >> 5;
  int wm = (wid >> 1) * 32, wn = (wid & 1) * 32;
  const bf16* Ap = q    + bh * 2048L * 512;
  const bf16* Bp = kmat + bh * 2048L * 512;
  if (tid < 128) sM[tid] = Mx[bh * 2048 + s0 + tid];
  if (tid < 64)  sG[tid] = G [bh * 2048 + t0 + tid];
  v8f acc[2][2] = {};
  for (int k0 = 0; k0 < 512; k0 += 32) {
    __syncthreads();
    stage_rows(lA, Ap, 512, s0, k0, 128, tid);  // q rows
    stage_rows(lB, Bp, 512, t0, k0, 64, tid);   // k rows are already the B^T rows
    stage_commit(tid);
    FragU a0, a1, b0, b1;
    lds_frag(lA, wm, lane, a0);  lds_frag(lA, wm + 16, lane, a1);
    lds_frag(lB, wn, lane, b0);  lds_frag(lB, wn + 16, lane, b1);
    acc[0][0] = wmma_bf16(a0, b0, acc[0][0]);
    acc[0][1] = wmma_bf16(a0, b1, acc[0][1]);
    acc[1][0] = wmma_bf16(a1, b0, acc[1][0]);
    acc[1][1] = wmma_bf16(a1, b1, acc[1][1]);
  }
  const float scale = 0.044194173824159216f;  // 512^-0.5
  int nl = lane & 15, hi = lane >> 4;
#pragma unroll
  for (int mi = 0; mi < 2; ++mi) {
#pragma unroll
    for (int r = 0; r < 8; ++r) {
      int  sl = wm + mi * 16 + r + 8 * hi;
      long sg = s0 + sl;
      float Ms = sM[sl];
      float rp = 0.f;
#pragma unroll
      for (int ni = 0; ni < 2; ++ni) {
        int  tl = wn + ni * 16 + nl;
        long tg = t0 + tl;
        float v = 0.f;
        if (tg <= sg) v = acc[mi][ni][r] * scale * expf(sG[tl] - Ms);  // exponent <= 0
        rp += v;
        Cb[(bh * 2048 + sg) * 2048 + tg] = (bf16)v;
      }
#pragma unroll
      for (int m = 8; m >= 1; m >>= 1) rp += __shfl_xor(rp, m, 32);  // reduce within 16-lane n-group
      if (nl == 0) atomicAdd(&rowsum[bh * 2048 + sg], rp);
    }
  }
}

// ---------------- O = (C @ V) / norm, triangular k-bound ----------------
__global__ void __launch_bounds__(256) k_pv(const bf16* __restrict__ Cb, const bf16* __restrict__ vmat,
                                            float* __restrict__ h, const float* __restrict__ nrm) {
  long bh = blockIdx.z;
  long s0 = (long)blockIdx.y * 128, n0 = (long)blockIdx.x * 64;
  __shared__ bf16 lA[128 * LPAD];
  __shared__ bf16 lB[64 * LPAD];
  __shared__ float sN[128];
  int tid = threadIdx.x, lane = tid & 31, wid = tid >> 5;
  int wm = (wid >> 1) * 32, wn = (wid & 1) * 32;
  const bf16* Ap = Cb   + bh * 2048L * 2048;
  const bf16* Bp = vmat + bh * 2048L * 512;
  if (tid < 128) sN[tid] = nrm[bh * 2048 + s0 + tid];
  v8f acc[2][2] = {};
  int kend = (int)s0 + 128;  // C is zero above the diagonal inside this range
  for (int k0 = 0; k0 < kend; k0 += 32) {
    __syncthreads();
    stage_rows(lA, Ap, 2048, s0, k0, 128, tid);
    stage_B64_trans(lB, Bp, 512, k0, n0, tid);
    stage_commit(tid);
    FragU a0, a1, b0, b1;
    lds_frag(lA, wm, lane, a0);  lds_frag(lA, wm + 16, lane, a1);
    lds_frag(lB, wn, lane, b0);  lds_frag(lB, wn + 16, lane, b1);
    acc[0][0] = wmma_bf16(a0, b0, acc[0][0]);
    acc[0][1] = wmma_bf16(a0, b1, acc[0][1]);
    acc[1][0] = wmma_bf16(a1, b0, acc[1][0]);
    acc[1][1] = wmma_bf16(a1, b1, acc[1][1]);
  }
  int nl = lane & 15, hi = lane >> 4;
#pragma unroll
  for (int mi = 0; mi < 2; ++mi)
#pragma unroll
    for (int ni = 0; ni < 2; ++ni)
#pragma unroll
      for (int r = 0; r < 8; ++r) {
        int sl = wm + mi * 16 + r + 8 * hi;
        h[(bh * 2048 + s0 + sl) * 512 + n0 + wn + ni * 16 + nl] = acc[mi][ni][r] / sN[sl];
      }
}

// ---------------- elementwise / small kernels ----------------
__global__ void k_f2bf(const float* __restrict__ in, bf16* __restrict__ out, long n) {
  long i = (long)blockIdx.x * 256 + threadIdx.x;
  if (i < n) out[i] = (bf16)in[i];
}

__global__ void k_conv(const float* __restrict__ xin, const float* __restrict__ cw,
                       const float* __restrict__ cb, float* __restrict__ act) {
  long i = (long)blockIdx.x * 256 + threadIdx.x;
  if (i >= 4096L * 2048) return;
  long t = i >> 11; int c = (int)(i & 2047); int s = (int)(t & 2047);
  float a = cb[c];
#pragma unroll
  for (int j = 0; j < 4; ++j) {
    int sp = s - 3 + j;
    if (sp >= 0) a += cw[j * 2048 + c] * xin[(t - 3 + j) * 4096 + c];  // x_mlstm = cols [0,2048)
  }
  float sg = 1.f / (1.f + expf(-a));
  act[t * 2048 + c] = a * sg;
}

__global__ void k_headwise(const float* __restrict__ act, const float* __restrict__ xin,
                           const float* __restrict__ qk_, const float* __restrict__ kk_,
                           const float* __restrict__ vk_,
                           bf16* __restrict__ qb, bf16* __restrict__ kb, bf16* __restrict__ vb) {
  long i = (long)blockIdx.x * 256 + threadIdx.x;
  if (i >= 4096L * 512) return;
  long t = i >> 9; int ph = (int)(i & 511);
  int c0 = ph * 4;
  float a4[4], xm4[4];
#pragma unroll
  for (int d = 0; d < 4; ++d) { a4[d] = act[t * 2048 + c0 + d]; xm4[d] = xin[t * 4096 + c0 + d]; }
  int h = ph >> 7;
  long b = t >> 11, s = t & 2047;
  long base = ((b * 4 + h) * 2048L + s) * 512 + (c0 - h * 512);
#pragma unroll
  for (int o = 0; o < 4; ++o) {
    float s1 = 0.f, s2 = 0.f, s3 = 0.f;
#pragma unroll
    for (int d = 0; d < 4; ++d) {
      s1 += qk_[(ph * 4 + o) * 4 + d] * a4[d];
      s2 += kk_[(ph * 4 + o) * 4 + d] * a4[d];
      s3 += vk_[(ph * 4 + o) * 4 + d] * xm4[d];
    }
    qb[base + o] = (bf16)s1; kb[base + o] = (bf16)s2; vb[base + o] = (bf16)s3;
  }
}

__global__ void __launch_bounds__(256) k_gates(const bf16* __restrict__ qb, const bf16* __restrict__ kb,
                                               const bf16* __restrict__ vb,
                                               const float* __restrict__ igw, const float* __restrict__ igb,
                                               const float* __restrict__ fgw, const float* __restrict__ fgb,
                                               float* __restrict__ ig, float* __restrict__ fgr) {
  __shared__ float red[256];
  long t = blockIdx.x; long b = t >> 11, s = t & 2047;
  int tid = threadIdx.x;
  float ia[4] = {0, 0, 0, 0}, fa[4] = {0, 0, 0, 0};
  for (int c = tid; c < 6144; c += 256) {
    int seg = c >> 11, cc = c & 2047, h = cc >> 9, e = cc & 511;
    const bf16* p = (seg == 0) ? qb : (seg == 1) ? kb : vb;
    float v = (float)p[((b * 4 + h) * 2048 + s) * 512 + e];
#pragma unroll
    for (int g = 0; g < 4; ++g) { ia[g] += v * igw[c * 4 + g]; fa[g] += v * fgw[c * 4 + g]; }
  }
  for (int g = 0; g < 8; ++g) {
    float v = (g < 4) ? ia[g] : fa[g - 4];
    __syncthreads();
    red[tid] = v; __syncthreads();
    for (int o = 128; o >= 1; o >>= 1) { if (tid < o) red[tid] += red[tid + o]; __syncthreads(); }
    if (tid == 0) {
      if (g < 4) ig [(b * 4 + g)       * 2048 + s] = red[0] + igb[g];
      else       fgr[(b * 4 + (g - 4)) * 2048 + s] = red[0] + fgb[g - 4];
    }
  }
}

// Per (b,h): F=cumsum(logsigmoid(fg)), G=ig-F, M=cummax(G)
__global__ void __launch_bounds__(256) k_scan(const float* __restrict__ ig, const float* __restrict__ fgr,
                                              float* __restrict__ F, float* __restrict__ G, float* __restrict__ Mx) {
  __shared__ float part[256];
  long bh = blockIdx.x; int tid = threadIdx.x;
  const float* fr = fgr + bh * 2048;
  const float* ip = ig  + bh * 2048;
  float loc[8], run = 0.f;
#pragma unroll
  for (int j = 0; j < 8; ++j) {
    float x = fr[tid * 8 + j];
    float lf = fminf(x, 0.f) - log1pf(expf(-fabsf(x)));
    run += lf; loc[j] = run;
  }
  part[tid] = run; __syncthreads();
  for (int off = 1; off < 256; off <<= 1) {
    float tv = (tid >= off) ? part[tid - off] : 0.f;
    __syncthreads(); part[tid] += tv; __syncthreads();
  }
  float excl = (tid > 0) ? part[tid - 1] : 0.f;
  float Gv[8], locm[8], runm = -3.0e38f;
#pragma unroll
  for (int j = 0; j < 8; ++j) {
    float Fv = loc[j] + excl;
    Gv[j] = ip[tid * 8 + j] - Fv;
    F[bh * 2048 + tid * 8 + j] = Fv;
    G[bh * 2048 + tid * 8 + j] = Gv[j];
    runm = fmaxf(runm, Gv[j]); locm[j] = runm;
  }
  __syncthreads();
  part[tid] = runm; __syncthreads();
  for (int off = 1; off < 256; off <<= 1) {
    float tv = (tid >= off) ? part[tid - off] : -3.0e38f;
    __syncthreads(); part[tid] = fmaxf(part[tid], tv); __syncthreads();
  }
  float exm = (tid > 0) ? part[tid - 1] : -3.0e38f;
#pragma unroll
  for (int j = 0; j < 8; ++j) Mx[bh * 2048 + tid * 8 + j] = fmaxf(locm[j], exm);
}

__global__ void k_norm(const float* __restrict__ rowsum, const float* __restrict__ F,
                       const float* __restrict__ Mx, float* __restrict__ nrm) {
  long i = (long)blockIdx.x * 256 + threadIdx.x;
  if (i >= 8L * 2048) return;
  float m = F[i] + Mx[i];
  nrm[i] = fmaxf(fabsf(rowsum[i]), expf(-m)) + 1e-6f;
}

// Headwise LayerNorm + out_norm + skip*act, * swish(z); emit bf16 for the down GEMM.
__global__ void __launch_bounds__(256) k_ln(const float* __restrict__ h, const float* __restrict__ onw,
                                            const float* __restrict__ skp, const float* __restrict__ act,
                                            const float* __restrict__ xin, bf16* __restrict__ hs) {
  __shared__ float r1[256], r2[256];
  long t = blockIdx.x; long b = t >> 11, s = t & 2047;
  int tid = threadIdx.x;
  for (int hh = 0; hh < 4; ++hh) {
    long base = ((b * 4 + hh) * 2048 + s) * 512;
    float v0 = h[base + tid], v1 = h[base + tid + 256];
    __syncthreads();
    r1[tid] = v0 + v1; r2[tid] = v0 * v0 + v1 * v1;
    __syncthreads();
    for (int o = 128; o >= 1; o >>= 1) {
      if (tid < o) { r1[tid] += r1[tid + o]; r2[tid] += r2[tid + o]; }
      __syncthreads();
    }
    float mean = r1[0] * (1.f / 512.f);
    float var  = r2[0] * (1.f / 512.f) - mean * mean;
    float rs   = rsqrtf(var + 1e-5f);
#pragma unroll
    for (int q2 = 0; q2 < 2; ++q2) {
      int e = tid + q2 * 256;
      float hv = q2 ? v1 : v0;
      float hn = (hv - mean) * rs;
      int c = hh * 512 + e;
      float ho = hn * onw[c] + skp[c] * act[t * 2048 + c];
      float z  = xin[t * 4096 + 2048 + c];
      float sw = z / (1.f + expf(-z));
      hs[t * 2048 + c] = (bf16)(ho * sw);
    }
  }
}

extern "C" void kernel_launch(void* const* d_in, const int* in_sizes, int n_in,
                              void* d_out, int out_size, void* d_ws, size_t ws_size,
                              hipStream_t stream) {
  (void)in_sizes; (void)n_in; (void)out_size; (void)ws_size;
  const float* x   = (const float*)d_in[0];
  const float* Wup = (const float*)d_in[1];
  const float* qk_ = (const float*)d_in[2];
  const float* kk_ = (const float*)d_in[3];
  const float* vk_ = (const float*)d_in[4];
  const float* cw  = (const float*)d_in[5];
  const float* cb  = (const float*)d_in[6];
  const float* igw = (const float*)d_in[7];
  const float* igb = (const float*)d_in[8];
  const float* fgw = (const float*)d_in[9];
  const float* fgb = (const float*)d_in[10];
  const float* onw = (const float*)d_in[11];
  const float* skp = (const float*)d_in[12];
  const float* Wdn = (const float*)d_in[13];
  float* out = (float*)d_out;

  char* w = (char*)d_ws;
  size_t off = 0;
  auto alloc = [&](size_t bytes) -> char* {
    char* p = w + off;
    off = (off + bytes + 255) & ~(size_t)255;
    return p;
  };
  bf16*  xbf    = (bf16*) alloc(4096L * 1024 * 2);
  bf16*  upbf   = (bf16*) alloc(1024L * 4096 * 2);
  bf16*  dnbf   = (bf16*) alloc(2048L * 1024 * 2);
  float* xinner = (float*)alloc(4096L * 4096 * 4);
  float* act    = (float*)alloc(4096L * 2048 * 4);
  bf16*  qb     = (bf16*) alloc(4096L * 2048 * 2);
  bf16*  kb     = (bf16*) alloc(4096L * 2048 * 2);
  bf16*  vb     = (bf16*) alloc(4096L * 2048 * 2);
  float* ig     = (float*)alloc(16384 * 4);
  float* fgr    = (float*)alloc(16384 * 4);
  float* F      = (float*)alloc(16384 * 4);
  float* G      = (float*)alloc(16384 * 4);
  float* Mx     = (float*)alloc(16384 * 4);
  float* rowsum = (float*)alloc(16384 * 4);
  float* nrm    = (float*)alloc(16384 * 4);
  bf16*  Cb     = (bf16*) alloc(8L * 2048 * 2048 * 2);
  float* hb     = (float*)alloc(4096L * 2048 * 4);
  bf16*  hs     = (bf16*) alloc(4096L * 2048 * 2);

  k_f2bf<<<(4096L * 1024 + 255) / 256, 256, 0, stream>>>(x,   xbf,  4096L * 1024);
  k_f2bf<<<(1024L * 4096 + 255) / 256, 256, 0, stream>>>(Wup, upbf, 1024L * 4096);
  k_f2bf<<<(2048L * 1024 + 255) / 256, 256, 0, stream>>>(Wdn, dnbf, 2048L * 1024);

  // x_inner = x @ Wup : M=4096 K=1024 N=4096
  k_gemm_bt<<<dim3(64, 32), 256, 0, stream>>>(xbf, upbf, xinner, 1024, 1024, 4096, 4096);

  k_conv<<<(4096L * 2048) / 256, 256, 0, stream>>>(xinner, cw, cb, act);
  k_headwise<<<(4096L * 512) / 256, 256, 0, stream>>>(act, xinner, qk_, kk_, vk_, qb, kb, vb);
  k_gates<<<4096, 256, 0, stream>>>(qb, kb, vb, igw, igb, fgw, fgb, ig, fgr);
  k_scan<<<8, 256, 0, stream>>>(ig, fgr, F, G, Mx);

  hipMemsetAsync(rowsum, 0, 16384 * 4, stream);
  k_qk<<<dim3(32, 16, 8), 256, 0, stream>>>(qb, kb, Cb, rowsum, G, Mx);
  k_norm<<<64, 256, 0, stream>>>(rowsum, F, Mx, nrm);
  k_pv<<<dim3(8, 16, 8), 256, 0, stream>>>(Cb, vb, hb, nrm);

  k_ln<<<4096, 256, 0, stream>>>(hb, onw, skp, act, xinner, hs);

  // out = h_state @ Wdown : M=4096 K=2048 N=1024
  k_gemm_bt<<<dim3(16, 32), 256, 0, stream>>>(hs, dnbf, out, 2048, 2048, 1024, 1024);
}
